// SwinTransformerBlock_73778948211083
// MI455X (gfx1250) — matrix operand order repaired and checked by reading, not compiled, not run on hardware
//
#include <hip/hip_runtime.h>
#include <math.h>

#if defined(__has_include)
#  if __has_include(<hip/amd_detail/amd_gfx1250_TDM.h>)
#    define ATH_TDM_6ARG 1
#  else
#    define ATH_TDM_6ARG 0
#  endif
#else
#  define ATH_TDM_6ARG 0
#endif

typedef unsigned short ushort_t;
typedef __attribute__((ext_vector_type(16))) __bf16 bf16x16;
typedef __attribute__((ext_vector_type(8)))  float  f32x8;
typedef __attribute__((ext_vector_type(4)))  float  f32x4;
typedef __attribute__((ext_vector_type(4)))  unsigned int u32x4;
typedef __attribute__((ext_vector_type(8)))  int    i32x8;
typedef __attribute__((ext_vector_type(4)))  int    i32x4;

#define CB    192
#define HH    56
#define WW_   56
#define BB    32
#define NHH   6
#define HD    32
#define WSZ   7
#define NWIN  49
#define NPAD  64
#define BWIN  (BB * 8 * 8)        /* 2048 windows */
#define MPAD  (BWIN * NPAD)       /* 131072 padded rows */
#define MROWS (BB * HH * WW_)     /* 100352 rows */
#define MLPH  768
#define SCALEQ 0.17677669529663689f
#define EPSV   1e-5f

// ---------------------------------------------------------------- helpers

__device__ __forceinline__ ushort_t f2bf(float f) {
  union { float f; unsigned u; } c; c.f = f;
  unsigned r = c.u + 0x7FFFu + ((c.u >> 16) & 1u);
  return (ushort_t)(r >> 16);
}

// ds_swizzle-based lane exchange (immediate offsets, no VALU address setup)
template <int MASK>
__device__ __forceinline__ float swz_xor_f(float x) {
  return __int_as_float(
      __builtin_amdgcn_ds_swizzle(__float_as_int(x), 0x1f | (MASK << 10)));
}
__device__ __forceinline__ float swz_bcast16_f(float x) {  // lane 0 / lane 16
  return __int_as_float(__builtin_amdgcn_ds_swizzle(__float_as_int(x), 0x10));
}
__device__ __forceinline__ float wave_sum32(float s) {
  s += swz_xor_f<1>(s); s += swz_xor_f<2>(s); s += swz_xor_f<4>(s);
  s += swz_xor_f<8>(s); s += swz_xor_f<16>(s);
  return s;
}

union FragU { bf16x16 v; u32x4 q[2]; };

__device__ __forceinline__ f32x8 wmma_bf16(const FragU& a, const FragU& b, f32x8 c) {
  return __builtin_amdgcn_wmma_f32_16x16x32_bf16(false, a.v, false, b.v,
                                                 (short)0, c, false, false);
}

// A fragment: 16x32 bf16 tile, row = lane%16, K run = 16 contiguous per lane half
__device__ __forceinline__ FragU load_a_g(const ushort_t* A, int ldk, int row,
                                          int k0, int lh) {
  FragU f;
  const ushort_t* p = A + (size_t)row * ldk + k0 + lh * 16;
  f.q[0] = *(const u32x4*)p;
  f.q[1] = *(const u32x4*)(p + 8);
  return f;
}
// B fragment: 32x16 bf16 tile, col = lane%16, K run = 16 contiguous per lane half
__device__ __forceinline__ FragU load_b_g(const ushort_t* Wt, int ldk, int col,
                                          int k0, int lh) {
  FragU f;
  const ushort_t* p = Wt + (size_t)col * ldk + k0 + lh * 16;
  f.q[0] = *(const u32x4*)p;
  f.q[1] = *(const u32x4*)(p + 8);
  return f;
}

// 1D TDM copy global -> LDS, nbytes multiple of 8
__device__ __forceinline__ void tdm_load_1d(unsigned lds_off, const void* gsrc,
                                            unsigned nbytes) {
  unsigned long long ga = (unsigned long long)(size_t)gsrc;
  unsigned n8 = nbytes >> 3;                 // 8-byte elements
  u32x4 g0;
  g0.x = 1u;                                 // count=1, user descriptor
  g0.y = lds_off;                            // LDS byte address
  g0.z = (unsigned)(ga & 0xFFFFFFFFull);     // global_addr[31:0]
  g0.w = (unsigned)((ga >> 32) & 0x01FFFFFFull) | (2u << 30); // [56:32] | type=2
  i32x8 g1;
  g1[0] = (int)(3u << 16);                          // data_size=8B
  g1[1] = (int)((n8 & 0xFFFFu) << 16);              // tensor_dim0[15:0]
  g1[2] = (int)(((n8 >> 16) & 0xFFFFu) | (1u << 16)); // dim0[31:16] | dim1=1
  g1[3] = (int)((n8 & 0xFFFFu) << 16);              // tile_dim0
  g1[4] = 1;                                        // tile_dim1=1, tile_dim2=0
  g1[5] = (int)n8;                                  // dim0_stride[31:0]
  g1[6] = (int)((n8 & 0xFFFFu) << 16);              // dim1_stride[15:0]
  g1[7] = (int)(n8 >> 16);                          // dim1_stride[47:16]
  i32x4 z4 = {0, 0, 0, 0};
#if ATH_TDM_6ARG
  i32x8 z8 = {0, 0, 0, 0, 0, 0, 0, 0};
  __builtin_amdgcn_tensor_load_to_lds(g0, g1, z4, z4, z8, 0);
#else
  __builtin_amdgcn_tensor_load_to_lds(g0, g1, z4, z4, 0);
#endif
}

// ---------------------------------------------------------------- kernels

__global__ __launch_bounds__(256) void cvt_bf16_k(const float* __restrict__ s,
                                                  ushort_t* __restrict__ d, int n) {
  int i = blockIdx.x * 256 + threadIdx.x;
  if (i < n) d[i] = f2bf(s[i]);
}

// Fused rel-pos bias + shifted-window mask, pre-swizzled to the WMMA C-fragment
// order: bm[cls][head][wave][tile][lane][g]  (8 contiguous floats per lane/tile)
__global__ __launch_bounds__(256) void biasmask_k(const float* __restrict__ bt,
                                                  float* __restrict__ bm) {
  int i = blockIdx.x * 256 + threadIdx.x;   // 4*NHH*4096 total
  int g    = i & 7;
  int lane = (i >> 3) & 31;
  int t    = (i >> 8) & 3;
  int wv   = (i >> 10) & 3;
  int hd   = (i >> 12) % NHH;
  int cls  = i / (NHH * 4096);
  int m = wv * 16 + g + (lane >> 4) * 8;
  int j = t * 16 + (lane & 15);
  float v;
  if (j >= NWIN) {
    v = -1e30f;
  } else {
    int mi = m / WSZ, mj = m % WSZ, ji = j / WSZ, jj = j % WSZ;
    int r = mi - ji + 6;  if (r < 0) r = 0;  if (r > 12) r = 12;
    int cc = mj - jj + 6; if (cc < 0) cc = 0; if (cc > 12) cc = 12;
    v = bt[(r * 13 + cc) * NHH + hd];
    int ch = cls >> 1, cw = cls & 1;
    int rm = (ch ? (mi < 4 ? 1 : 2) : 0) * 3 + (cw ? (mj < 4 ? 1 : 2) : 0);
    int rj = (ch ? (ji < 4 ? 1 : 2) : 0) * 3 + (cw ? (jj < 4 ? 1 : 2) : 0);
    if (rm != rj) v -= 100.0f;
  }
  bm[i] = v;
}

// LN1 + cyclic shift + window partition -> bf16 [MPAD, CB], pad rows zeroed
__global__ __launch_bounds__(256) void ln1_shift_k(const float* __restrict__ x,
                                                   const float* __restrict__ gw,
                                                   const float* __restrict__ bw,
                                                   ushort_t* __restrict__ xwb) {
  int lane = threadIdx.x & 31, wv = threadIdx.x >> 5;
  int row = blockIdx.x * 8 + wv;
  ushort_t* dst = xwb + (size_t)row * CB;
  int win = row >> 6, n = row & 63;
  if (n >= NWIN) {
#pragma unroll
    for (int t = 0; t < 6; ++t) dst[t * 32 + lane] = 0;
    return;
  }
  int b = win >> 6, wh = (win >> 3) & 7, wc = win & 7;
  int hI = wh * WSZ + n / WSZ + 3; if (hI >= HH) hI -= HH;
  int wI = wc * WSZ + n % WSZ + 3; if (wI >= WW_) wI -= WW_;
  const float* src = x + ((size_t)b * (HH * WW_) + hI * WW_ + wI) * CB;
  float v[6]; float s = 0.f;
#pragma unroll
  for (int t = 0; t < 6; ++t) { v[t] = src[t * 32 + lane]; s += v[t]; }
  s = wave_sum32(s);
  float mu = s * (1.0f / CB), q = 0.f;
#pragma unroll
  for (int t = 0; t < 6; ++t) { float d = v[t] - mu; q += d * d; }
  q = wave_sum32(q);
  float rs = rsqrtf(q * (1.0f / CB) + EPSV);
#pragma unroll
  for (int t = 0; t < 6; ++t) {
    int c = t * 32 + lane;
    dst[c] = f2bf((v[t] - mu) * rs * gw[c] + bw[c]);
  }
}

// LN2 on f32 rows -> bf16
__global__ __launch_bounds__(256) void ln2_k(const float* __restrict__ xm,
                                             const float* __restrict__ gw,
                                             const float* __restrict__ bw,
                                             ushort_t* __restrict__ o) {
  int lane = threadIdx.x & 31, wv = threadIdx.x >> 5;
  int row = blockIdx.x * 8 + wv;
  const float* src = xm + (size_t)row * CB;
  ushort_t* dst = o + (size_t)row * CB;
  float v[6]; float s = 0.f;
#pragma unroll
  for (int t = 0; t < 6; ++t) { v[t] = src[t * 32 + lane]; s += v[t]; }
  s = wave_sum32(s);
  float mu = s * (1.0f / CB), q = 0.f;
#pragma unroll
  for (int t = 0; t < 6; ++t) { float d = v[t] - mu; q += d * d; }
  q = wave_sum32(q);
  float rs = rsqrtf(q * (1.0f / CB) + EPSV);
#pragma unroll
  for (int t = 0; t < 6; ++t) {
    int c = t * 32 + lane;
    dst[c] = f2bf((v[t] - mu) * rs * gw[c] + bw[c]);
  }
}

// Generic bf16 WMMA GEMM, block tile 256x64 (8 waves; wave tile 32x64 = 8 accs,
// 8 wmma per k-step from 2 A-frags + 4 B-frags).
// MODE 0: QKV scatter  1: proj+reverse+roll+residual  2: FC1+GELU  3: FC2+residual
template <int MODE>
__global__ __launch_bounds__(256) void gemm_k(
    const ushort_t* __restrict__ A, const ushort_t* __restrict__ Wt, int K,
    const float* __restrict__ bias, const float* __restrict__ resid,
    float* __restrict__ outF, ushort_t* __restrict__ outB,
    ushort_t* __restrict__ qo, ushort_t* __restrict__ ko,
    ushort_t* __restrict__ vo) {
  const int lane = threadIdx.x & 31, wv = threadIdx.x >> 5;
  const int lh = lane >> 4, nl = lane & 15;
  const int r0 = blockIdx.y * 256 + wv * 32;
  const int c0 = blockIdx.x * 64;

  f32x8 zero = {0.f, 0.f, 0.f, 0.f, 0.f, 0.f, 0.f, 0.f};
  f32x8 acc[2][4];
#pragma unroll
  for (int rt = 0; rt < 2; ++rt)
#pragma unroll
    for (int ct = 0; ct < 4; ++ct) acc[rt][ct] = zero;

  for (int k0 = 0; k0 < K; k0 += 32) {
    FragU fa0 = load_a_g(A, K, r0 + nl, k0, lh);
    FragU fa1 = load_a_g(A, K, r0 + 16 + nl, k0, lh);
#pragma unroll
    for (int ct = 0; ct < 4; ++ct) {
      FragU fb = load_b_g(Wt, K, c0 + ct * 16 + nl, k0, lh);
      acc[0][ct] = wmma_bf16(fa0, fb, acc[0][ct]);
      acc[1][ct] = wmma_bf16(fa1, fb, acc[1][ct]);
    }
  }

#pragma unroll
  for (int rt = 0; rt < 2; ++rt)
#pragma unroll
    for (int ct = 0; ct < 4; ++ct) {
#pragma unroll
      for (int g = 0; g < 8; ++g) {
        float v = acc[rt][ct][g];
        int m = r0 + rt * 16 + g + lh * 8;
        int c = c0 + ct * 16 + nl;
        if (MODE == 0) {
          int sl = c / CB, hd = (c % CB) >> 5, e = c & 31;
          int win = m >> 6, n = m & 63;
          size_t bh = (size_t)(win * NHH + hd);
          if (sl == 0)      qo[(bh * NPAD + n) * HD + e] = f2bf(v * SCALEQ);
          else if (sl == 1) ko[(bh * NPAD + n) * HD + e] = f2bf(v);
          else              vo[(bh * HD + e) * NPAD + n] = f2bf(v);
        } else if (MODE == 1) {
          int win = m >> 6, n = m & 63;
          if (n < NWIN) {
            int b = win >> 6, wh = (win >> 3) & 7, wc = win & 7;
            int hI = wh * WSZ + n / WSZ + 3; if (hI >= HH) hI -= HH;
            int wI = wc * WSZ + n % WSZ + 3; if (wI >= WW_) wI -= WW_;
            size_t idx = ((size_t)b * (HH * WW_) + hI * WW_ + wI) * CB + c;
            outF[idx] = resid[idx] + v + bias[c];
          }
        } else if (MODE == 2) {
          float t = v + bias[c];
          t = 0.5f * t * (1.0f + erff(t * 0.70710678118654752f));
          outB[(size_t)m * MLPH + c] = f2bf(t);
        } else {
          size_t idx = (size_t)m * CB + c;
          outF[idx] = outF[idx] + v + bias[c];
        }
      }
    }
}

// Fused windowed attention: one block (4 waves) per (window, head).
// LDS element offsets into one shared block (byte offsets = 2x):
//   sK  [0,     2048)  k tile [j,e]        (TDM target @ byte 0)
//   sV  [2048,  5120)  v tile [e,j] rows 0..31 = V (TDM @ byte 4096),
//                      row 32 = ones (sum column), rows 33..47 = 0
//   sP  [5120,  9216)  per-wave unnormalized softmax rows (bf16)
#define SK_OFF 0
#define SV_OFF 2048
#define SP_OFF 5120
__global__ __launch_bounds__(128) void attn_k(const ushort_t* __restrict__ qb,
                                              const ushort_t* __restrict__ kb,
                                              const ushort_t* __restrict__ vtb,
                                              const float* __restrict__ bm,
                                              ushort_t* __restrict__ awb) {
  __shared__ ushort_t smem[9216];
  const int blk = blockIdx.x;
  const int win = blk / NHH, hd = blk - win * NHH;
  const int tid = threadIdx.x;
  const int lane = tid & 31, wv = tid >> 5;
  const int lh = lane >> 4, nl = lane & 15;
  const size_t bh = (size_t)(win * NHH + hd);

  if (wv == 0) {  // TDM async copies into LDS, then drain TENSORcnt
    tdm_load_1d(0u, kb + bh * (NPAD * HD), NPAD * HD * 2);
    tdm_load_1d(4096u, vtb + bh * (HD * NPAD), HD * NPAD * 2);
    __builtin_amdgcn_s_wait_tensorcnt(0);
  }
  // rows 32..47 of the V tile: row 32 = 1.0 (row-sum column), rest 0
#pragma unroll
  for (int u = 0; u < 8; ++u) {
    int o2 = tid * 8 + u;
    smem[SV_OFF + 2048 + o2] = (o2 < 64) ? (ushort_t)0x3F80 : (ushort_t)0;
  }
  __syncthreads();

  const int r0 = wv * 16;
  FragU fq = load_a_g(qb + (bh * NPAD + r0) * HD, HD, nl, 0, lh);

  f32x8 zero = {0.f, 0.f, 0.f, 0.f, 0.f, 0.f, 0.f, 0.f};
  f32x8 S[4];
#pragma unroll
  for (int t = 0; t < 4; ++t) S[t] = zero;
#pragma unroll
  for (int t = 0; t < 4; ++t) {
    FragU fk;
    const ushort_t* p = &smem[SK_OFF + (t * 16 + nl) * HD + lh * 16];
    fk.q[0] = *(const u32x4*)p;
    fk.q[1] = *(const u32x4*)(p + 8);
    S[t] = wmma_bf16(fq, fk, S[t]);
  }

  // fused bias+mask (pre-swizzled, 8 contiguous floats per lane/tile)
  const int wh = (win >> 3) & 7, wc = win & 7;
  const int cls = ((wh == 7) ? 2 : 0) | ((wc == 7) ? 1 : 0);
  const float* bmp =
      bm + ((((size_t)cls * NHH + hd) * 4 + wv) * 4) * 256 + lane * 8;
  float sc[4][8];
#pragma unroll
  for (int t = 0; t < 4; ++t) {
    const float* p = bmp + t * 256;
    f32x4 b0 = *(const f32x4*)p;
    f32x4 b1 = *(const f32x4*)(p + 4);
#pragma unroll
    for (int g = 0; g < 8; ++g) sc[t][g] = S[t][g] + (g < 4 ? b0[g & 3] : b1[g & 3]);
  }

  // row max via ds_swizzle xor within 16-lane groups, then unnormalized exp
  float mx[8];
#pragma unroll
  for (int g = 0; g < 8; ++g) {
    float m0 = fmaxf(fmaxf(sc[0][g], sc[1][g]), fmaxf(sc[2][g], sc[3][g]));
    m0 = fmaxf(m0, swz_xor_f<1>(m0));
    m0 = fmaxf(m0, swz_xor_f<2>(m0));
    m0 = fmaxf(m0, swz_xor_f<4>(m0));
    m0 = fmaxf(m0, swz_xor_f<8>(m0));
    mx[g] = m0;
  }
  ushort_t* Pw = &smem[SP_OFF + wv * 16 * NPAD];
#pragma unroll
  for (int t = 0; t < 4; ++t)
#pragma unroll
    for (int g = 0; g < 8; ++g)
      Pw[(g + lh * 8) * NPAD + t * 16 + nl] = f2bf(__expf(sc[t][g] - mx[g]));
  __syncthreads();

  // O = P @ [V | ones]: tile 2 col 0 accumulates the softmax row-sums
  f32x8 O[3];
#pragma unroll
  for (int t = 0; t < 3; ++t) O[t] = zero;
#pragma unroll
  for (int k0 = 0; k0 < NPAD; k0 += 32) {
    FragU fp;
    const ushort_t* pp = &Pw[nl * NPAD + k0 + lh * 16];
    fp.q[0] = *(const u32x4*)pp;
    fp.q[1] = *(const u32x4*)(pp + 8);
#pragma unroll
    for (int t = 0; t < 3; ++t) {
      FragU fv;
      const ushort_t* pv = &smem[SV_OFF + (t * 16 + nl) * NPAD + k0 + lh * 16];
      fv.q[0] = *(const u32x4*)pv;
      fv.q[1] = *(const u32x4*)(pv + 8);
      O[t] = wmma_bf16(fp, fv, O[t]);
    }
  }
#pragma unroll
  for (int g = 0; g < 8; ++g) {
    float inv = 1.0f / swz_bcast16_f(O[2][g]);  // row-sum from col 32 (lane 0/16)
    int m = r0 + g + lh * 8;
    size_t base = ((size_t)win * NPAD + m) * CB + hd * 32;
    awb[base + nl]      = f2bf(O[0][g] * inv);
    awb[base + 16 + nl] = f2bf(O[1][g] * inv);
  }
}

// ---------------------------------------------------------------- launch

extern "C" void kernel_launch(void* const* d_in, const int* in_sizes, int n_in,
                              void* d_out, int out_size, void* d_ws, size_t ws_size,
                              hipStream_t stream) {
  const float* x      = (const float*)d_in[0];
  const float* n1g    = (const float*)d_in[1];
  const float* n1b    = (const float*)d_in[2];
  const float* qkv_w  = (const float*)d_in[3];
  const float* rbt    = (const float*)d_in[4];
  const float* proj_w = (const float*)d_in[5];
  const float* proj_b = (const float*)d_in[6];
  const float* n2g    = (const float*)d_in[7];
  const float* n2b    = (const float*)d_in[8];
  const float* fc1_w  = (const float*)d_in[9];
  const float* fc1_b  = (const float*)d_in[10];
  const float* fc2_w  = (const float*)d_in[11];
  const float* fc2_b  = (const float*)d_in[12];
  float* out = (float*)d_out;

  char* ws = (char*)d_ws;
  size_t off = 0;
  auto take = [&](size_t bytes) {
    char* p = ws + off;
    off += (bytes + 255) & ~(size_t)255;
    return p;
  };
  ushort_t* qkv_wb  = (ushort_t*)take((size_t)3 * CB * CB * 2);
  ushort_t* proj_wb = (ushort_t*)take((size_t)CB * CB * 2);
  ushort_t* fc1_wb  = (ushort_t*)take((size_t)MLPH * CB * 2);
  ushort_t* fc2_wb  = (ushort_t*)take((size_t)CB * MLPH * 2);
  float*    bm      = (float*)take((size_t)4 * NHH * NPAD * NPAD * 4);
  ushort_t* xwb = (ushort_t*)take((size_t)MPAD * CB * 2);
  ushort_t* qb  = (ushort_t*)take((size_t)BWIN * NHH * NPAD * HD * 2);
  ushort_t* kb  = (ushort_t*)take((size_t)BWIN * NHH * NPAD * HD * 2);
  ushort_t* vtb = (ushort_t*)take((size_t)BWIN * NHH * HD * NPAD * 2);
  ushort_t* awb = (ushort_t*)take((size_t)MPAD * CB * 2);
  ushort_t* x2b = (ushort_t*)take((size_t)MROWS * CB * 2);
  ushort_t* hb  = (ushort_t*)take((size_t)MROWS * MLPH * 2);

  int nq = 3 * CB * CB, np = CB * CB, n1 = MLPH * CB, n2 = CB * MLPH;
  cvt_bf16_k<<<(nq + 255) / 256, 256, 0, stream>>>(qkv_w, qkv_wb, nq);
  cvt_bf16_k<<<(np + 255) / 256, 256, 0, stream>>>(proj_w, proj_wb, np);
  cvt_bf16_k<<<(n1 + 255) / 256, 256, 0, stream>>>(fc1_w, fc1_wb, n1);
  cvt_bf16_k<<<(n2 + 255) / 256, 256, 0, stream>>>(fc2_w, fc2_wb, n2);
  biasmask_k<<<(4 * NHH * NPAD * NPAD) / 256, 256, 0, stream>>>(rbt, bm);

  ln1_shift_k<<<MPAD / 8, 256, 0, stream>>>(x, n1g, n1b, xwb);

  gemm_k<0><<<dim3(3 * CB / 64, MPAD / 256), 256, 0, stream>>>(
      xwb, qkv_wb, CB, nullptr, nullptr, nullptr, nullptr, qb, kb, vtb);

  attn_k<<<BWIN * NHH, 128, 0, stream>>>(qb, kb, vtb, bm, awb);

  gemm_k<1><<<dim3(CB / 64, MPAD / 256), 256, 0, stream>>>(
      awb, proj_wb, CB, proj_b, x, out, nullptr, nullptr, nullptr, nullptr);

  ln2_k<<<MROWS / 8, 256, 0, stream>>>(out, n2g, n2b, x2b);

  gemm_k<2><<<dim3(MLPH / 64, MROWS / 256), 256, 0, stream>>>(
      x2b, fc1_wb, CB, fc1_b, nullptr, nullptr, hb, nullptr, nullptr, nullptr);

  gemm_k<3><<<dim3(CB / 64, MROWS / 256), 256, 0, stream>>>(
      hb, fc2_wb, MLPH, fc2_b, nullptr, out, nullptr, nullptr, nullptr, nullptr);
}
